// ChannelAttention_50362786513234
// MI455X (gfx1250) — matrix operand order repaired and checked by reading, not compiled, not run on hardware
//
#include <hip/hip_runtime.h>
#include <hip/hip_bf16.h>

// CDNA5 (gfx1250) wave32 WMMA types
typedef __attribute__((ext_vector_type(16))) __bf16 v16bf;
typedef __attribute__((ext_vector_type(8)))  float  v8f;
typedef __attribute__((ext_vector_type(4)))  __bf16 bf16x4;
typedef __attribute__((ext_vector_type(2)))  __bf16 bf16x2;

#define BATCH   16
#define CC      64
#define HWN     65536
#define NCHUNK  32
#define CHUNK   (HWN / NCHUNK)     // 2048 n-values per split-K chunk
#define KSTEP   128                // K slab staged in LDS per iteration
#define LROW    (KSTEP + 8)        // 136 elems = 272B row: 16B-aligned rows, staggered banks
#define NBLK    128                // n-columns per block in GEMM2
#define QROW    (NBLK + 4)
#define PROW    (CC + 4)

// ---------------------------------------------------------------------------
// Kernel 1: split-K GEMM1 partials.  S_part[b][chunk][c][d] =
//   sum_{n in chunk} x[b][c*HW+n] * x[b][n*64+d]    (bf16 WMMA, f32 accum)
// ---------------------------------------------------------------------------
__global__ __launch_bounds__(256)
void ca_gemm1_partial(const float* __restrict__ x, float* __restrict__ s_part) {
    __shared__ __bf16 As[CC][LROW];   // q slab: As[c][kk]
    __shared__ __bf16 Bs[CC][LROW];   // k slab transposed: Bs[d][kk]

    const int b     = blockIdx.x / NCHUNK;
    const int chunk = blockIdx.x % NCHUNK;
    const float* xb = x + (size_t)b * CC * HWN;

    const int tid  = threadIdx.x;
    const int lane = tid & 31;
    const int wave = tid >> 5;           // 0..7
    const int ci   = wave & 3;           // c-tile (0..3)
    const int di0  = (wave >> 2) * 2;    // d-tiles di0, di0+1
    const int mrow = lane & 15;
    const int hi   = lane >> 4;

    // staging decomposition for the k slab: 2 d-columns x 2 n-rows per step
    const int d2  = (tid & 31) * 2;      // even d column pair (0..62)
    const int ng  = tid >> 5;            // 0..7

    v8f acc0 = {};
    v8f acc1 = {};

    for (int it = 0; it < CHUNK / KSTEP; ++it) {
        const int n0 = chunk * CHUNK + it * KSTEP;

        // stage q slab: 64 rows (stride HW) x KSTEP cols, fp32 -> packed bf16x4
        for (int idx = tid; idx < CC * (KSTEP / 4); idx += 256) {
            const int row = idx / (KSTEP / 4);
            const int c4  = idx % (KSTEP / 4);
            const float4 v = *(const float4*)(xb + (size_t)row * HWN + n0 + c4 * 4);
            bf16x4 pk;
            pk.x = (__bf16)v.x; pk.y = (__bf16)v.y;
            pk.z = (__bf16)v.z; pk.w = (__bf16)v.w;
            *(bf16x4*)(&As[row][c4 * 4]) = pk;        // one b64 LDS store
        }
        // stage k slab transposed: global float2 (coalesced 256B/wave),
        // packed bf16x2 b32 stores into rows d2 / d2+1
        const float* kb = xb + (size_t)n0 * CC;
        #pragma unroll
        for (int m = 0; m < (KSTEP / 2) / 8; ++m) {   // 8 n-pairs per thread
            const int np = ng + m * 8;                // n-pair index 0..63
            const float2 r0 = *(const float2*)(kb + (size_t)(2 * np)     * CC + d2);
            const float2 r1 = *(const float2*)(kb + (size_t)(2 * np + 1) * CC + d2);
            bf16x2 p0; p0.x = (__bf16)r0.x; p0.y = (__bf16)r1.x;
            bf16x2 p1; p1.x = (__bf16)r0.y; p1.y = (__bf16)r1.y;
            *(bf16x2*)(&Bs[d2][2 * np])     = p0;     // one b32 LDS store
            *(bf16x2*)(&Bs[d2 + 1][2 * np]) = p1;     // one b32 LDS store
        }
        __syncthreads();

        #pragma unroll
        for (int ks = 0; ks < KSTEP / 32; ++ks) {
            // A fragment (ISA 16-bit 16x32 layout): lane-group k split 0-7/16-23 vs 8-15/24-31
            v16bf a;
            {
                const __bf16* ap = &As[ci * 16 + mrow][ks * 32 + hi * 8];
                #pragma unroll
                for (int i = 0; i < 8; ++i) a[i]     = ap[i];
                #pragma unroll
                for (int i = 0; i < 8; ++i) a[8 + i] = ap[16 + i];
            }
            // B fragments: same k split as A: 0-7/16-23 vs 8-15/24-31
            v16bf b0, b1;
            {
                const __bf16* bp = &Bs[di0 * 16 + mrow][ks * 32 + hi * 8];
                #pragma unroll
                for (int i = 0; i < 16; ++i) b0[i] = bp[i < 8 ? i : i + 8];
                const __bf16* bq = &Bs[(di0 + 1) * 16 + mrow][ks * 32 + hi * 8];
                #pragma unroll
                for (int i = 0; i < 16; ++i) b1[i] = bq[i < 8 ? i : i + 8];
            }
            acc0 = __builtin_amdgcn_wmma_f32_16x16x32_bf16(false, a, false, b0, (short)0, acc0, false, false);
            acc1 = __builtin_amdgcn_wmma_f32_16x16x32_bf16(false, a, false, b1, (short)0, acc1, false, false);
        }
        __syncthreads();
    }

    // write partial 16x16 tiles: D layout -> lane holds col (lane%16), rows 8*hi..
    float* sp = s_part + (((size_t)b * NCHUNK + chunk) * CC * CC);
    const int col0  = di0 * 16 + mrow;
    const int rbase = ci * 16 + hi * 8;
    #pragma unroll
    for (int r = 0; r < 8; ++r) {
        sp[(rbase + r) * CC + col0]      = acc0[r];
        sp[(rbase + r) * CC + col0 + 16] = acc1[r];
    }
}

// ---------------------------------------------------------------------------
// Kernel 2: deterministic split-K reduce + row softmax -> P[b][c][d]
// ---------------------------------------------------------------------------
__global__ __launch_bounds__(64)
void ca_softmax(const float* __restrict__ s_part, float* __restrict__ p) {
    const int b = blockIdx.x / CC;
    const int c = blockIdx.x % CC;
    const int d = threadIdx.x;

    float s = 0.f;
    for (int ch = 0; ch < NCHUNK; ++ch)
        s += s_part[(((size_t)b * NCHUNK + ch) * CC + c) * CC + d];

    __shared__ float row[CC];
    __shared__ float red[2];
    row[d] = s;
    __syncthreads();
    if (d == 0) {
        float m = row[0];
        for (int i = 1; i < CC; ++i) m = fmaxf(m, row[i]);
        float sum = 0.f;
        for (int i = 0; i < CC; ++i) sum += __expf(row[i] - m);
        red[0] = m;
        red[1] = sum;
    }
    __syncthreads();
    p[((size_t)b * CC + c) * CC + d] = __expf(s - red[0]) / red[1];
}

// ---------------------------------------------------------------------------
// Kernel 3: out[b][c][n] = sum_d P[b][c][d] * x[b][d*HW+n] + x[b][c*HW+n]
// fp32 x slab in LDS feeds both bf16 B-fragments and the exact residual.
// ---------------------------------------------------------------------------
__global__ __launch_bounds__(256)
void ca_gemm2(const float* __restrict__ x, const float* __restrict__ p,
              float* __restrict__ out) {
    __shared__ float Qs[CC][QROW];   // x slab (rows d == rows c), fp32
    __shared__ float Ps[CC][PROW];   // softmax matrix, fp32

    const int b  = blockIdx.x / (HWN / NBLK);
    const int nb = blockIdx.x % (HWN / NBLK);
    const int n0 = nb * NBLK;
    const float* xb = x + (size_t)b * CC * HWN;

    const int tid  = threadIdx.x;
    const int lane = tid & 31;
    const int wave = tid >> 5;        // n-tile index 0..7
    const int mrow = lane & 15;
    const int hi   = lane >> 4;

    for (int idx = tid; idx < CC * (NBLK / 4); idx += 256) {
        const int row = idx / (NBLK / 4);
        const int c4  = idx % (NBLK / 4);
        const float4 v = *(const float4*)(xb + (size_t)row * HWN + n0 + c4 * 4);
        Qs[row][c4 * 4 + 0] = v.x;
        Qs[row][c4 * 4 + 1] = v.y;
        Qs[row][c4 * 4 + 2] = v.z;
        Qs[row][c4 * 4 + 3] = v.w;
    }
    const float* pb = p + (size_t)b * CC * CC;
    for (int idx = tid; idx < CC * CC / 4; idx += 256) {
        const int row = idx / (CC / 4);
        const int c4  = idx % (CC / 4);
        const float4 v = *(const float4*)(pb + row * CC + c4 * 4);
        Ps[row][c4 * 4 + 0] = v.x;
        Ps[row][c4 * 4 + 1] = v.y;
        Ps[row][c4 * 4 + 2] = v.z;
        Ps[row][c4 * 4 + 3] = v.w;
    }
    __syncthreads();

    // Hoisted A fragments: af[ks][ci] from P (fp32 -> bf16)
    v16bf af[2][4];
    #pragma unroll
    for (int ks = 0; ks < 2; ++ks)
        #pragma unroll
        for (int ci = 0; ci < 4; ++ci) {
            const float* ap = &Ps[ci * 16 + mrow][ks * 32 + hi * 8];
            #pragma unroll
            for (int i = 0; i < 8; ++i) af[ks][ci][i]     = (__bf16)ap[i];
            #pragma unroll
            for (int i = 0; i < 8; ++i) af[ks][ci][8 + i] = (__bf16)ap[16 + i];
        }

    const int ncol = wave * 16 + mrow;
    v8f acc[4] = {{}, {}, {}, {}};
    #pragma unroll
    for (int ks = 0; ks < 2; ++ks) {
        v16bf bf;
        #pragma unroll
        for (int i = 0; i < 16; ++i)
            bf[i] = (__bf16)Qs[ks * 32 + hi * 8 + (i < 8 ? i : i + 8)][ncol];
        #pragma unroll
        for (int ci = 0; ci < 4; ++ci)
            acc[ci] = __builtin_amdgcn_wmma_f32_16x16x32_bf16(false, af[ks][ci], false, bf, (short)0, acc[ci], false, false);
    }

    float* ob = out + (size_t)b * CC * HWN;
    #pragma unroll
    for (int ci = 0; ci < 4; ++ci) {
        const int rbase = ci * 16 + hi * 8;
        #pragma unroll
        for (int r = 0; r < 8; ++r) {
            const int c = rbase + r;
            ob[(size_t)c * HWN + n0 + ncol] = acc[ci][r] + Qs[c][ncol];
        }
    }
}

// ---------------------------------------------------------------------------
extern "C" void kernel_launch(void* const* d_in, const int* in_sizes, int n_in,
                              void* d_out, int out_size, void* d_ws, size_t ws_size,
                              hipStream_t stream) {
    (void)in_sizes; (void)n_in; (void)out_size; (void)ws_size;
    const float* x   = (const float*)d_in[0];
    float*       out = (float*)d_out;

    // workspace layout: S_part [B][NCHUNK][64][64] f32, then P [B][64][64] f32
    float* s_part = (float*)d_ws;
    float* p      = s_part + (size_t)BATCH * NCHUNK * CC * CC;

    ca_gemm1_partial<<<dim3(BATCH * NCHUNK), dim3(256), 0, stream>>>(x, s_part);
    ca_softmax<<<dim3(BATCH * CC), dim3(64), 0, stream>>>(s_part, p);
    ca_gemm2<<<dim3(BATCH * (HWN / NBLK)), dim3(256), 0, stream>>>(x, p, out);
}
